// SparseCosineKernel_48120813584931
// MI455X (gfx1250) — matrix operand order repaired and checked by reading, not compiled, not run on hardware
//
#include <hip/hip_runtime.h>

typedef __attribute__((ext_vector_type(16))) __bf16 v16bf;
typedef __attribute__((ext_vector_type(8)))  float  v8f;

#define DIM 64          // feature dimension D (fixed by the reference)
#define COLS_PER_WG 128 // 4 waves * 32 columns

// first index i with a[i] >= key (a sorted ascending)
__device__ __forceinline__ int lower_bound_i(const int* __restrict__ a, int n, int key) {
    int lo = 0, hi = n;
    while (lo < hi) {
        int mid = (lo + hi) >> 1;
        if (a[mid] < key) lo = mid + 1; else hi = mid;
    }
    return lo;
}

// Load 2x8 consecutive floats (row+k0..+7, row+k1..+7) and split each into
// bf16 hi + bf16 lo residual (x ~= hi + lo), packed per the WMMA 16-bit
// operand register layout (vector element e <-> e-th 16-bit slot).
__device__ __forceinline__ void load_split16(const float* __restrict__ row,
                                             int k0, int k1,
                                             v16bf& h, v16bf& l) {
    const float4* p0 = reinterpret_cast<const float4*>(row + k0);
    const float4* p1 = reinterpret_cast<const float4*>(row + k1);
    float4 a0 = p0[0], a1 = p0[1], b0 = p1[0], b1 = p1[1];
    float f[16] = { a0.x, a0.y, a0.z, a0.w,  a1.x, a1.y, a1.z, a1.w,
                    b0.x, b0.y, b0.z, b0.w,  b1.x, b1.y, b1.z, b1.w };
#pragma unroll
    for (int e = 0; e < 16; ++e) {
        __bf16 hh = (__bf16)f[e];
        h[e] = hh;
        l[e] = (__bf16)(f[e] - (float)hh);
    }
}

#define WMMA_BF16(A, B, C) \
    __builtin_amdgcn_wmma_f32_16x16x32_bf16(false, (A), false, (B), (short)0, (C), false, false)

__global__ __launch_bounds__(128)
void sparse_cosine_kmat_kernel(const float* __restrict__ PS,    // [N, 64]
                               const float* __restrict__ SP,    // [M, 64]
                               const int*   __restrict__ zs,    // [N]
                               const int*   __restrict__ batch, // [N] sorted
                               const int*   __restrict__ psp,   // [M]
                               float*       __restrict__ out,   // [NMOL, M]
                               int N, int M) {
    const int b    = blockIdx.x;           // molecule / segment id
    const int wave = threadIdx.x >> 5;     // 4 waves per workgroup
    const int lane = threadIdx.x & 31;     // wave32
    const int colBase = blockIdx.y * COLS_PER_WG + wave * 32;  // 2 subtiles/wave
    if (colBase >= M) return;              // wave-uniform exit; no barriers used

    const int hi   = lane >> 4;            // lane half (C rows r vs r+8; B K halves)
    const int nrel = lane & 15;            // column within a 16-col subtile
    const int col0 = colBase + nrel;       // subtile 0
    const int col1 = colBase + 16 + nrel;  // subtile 1
    const int c0Cl = (col0 < M) ? col0 : (M - 1);
    const int c1Cl = (col1 < M) ? col1 : (M - 1);
    const int sc0  = (col0 < M) ? psp[c0Cl] : -2;   // sentinels never match z
    const int sc1  = (col1 < M) ? psp[c1Cl] : -2;

    // Segment bounds of molecule b in the sorted batch array.
    const int rowStart = lower_bound_i(batch, N, b);
    const int rowEnd   = lower_bound_i(batch, N, b + 1);

    // B = SP^T tiles (64x16 each), bf16 hi/lo split, register-resident across
    // the whole row loop. B (32x16) k order used here (contiguous, same as A): lanes 0-15 hold
    // K=0..15, lanes 16-31 hold K=16..31 (consecutive per lane).
    const float* br0 = SP + (size_t)c0Cl * DIM;
    const float* br1 = SP + (size_t)c1Cl * DIM;
    v16bf b0h_t0, b0l_t0, b1h_t0, b1l_t0;   // tile0: K chunks 0..31, 32..63
    v16bf b0h_t1, b0l_t1, b1h_t1, b1l_t1;   // tile1
    load_split16(br0,      16 * hi, 16 * hi + 8, b0h_t0, b0l_t0);
    load_split16(br0 + 32, 16 * hi, 16 * hi + 8, b1h_t0, b1l_t0);
    load_split16(br1,      16 * hi, 16 * hi + 8, b0h_t1, b0l_t1);
    load_split16(br1 + 32, 16 * hi, 16 * hi + 8, b1h_t1, b1l_t1);

    float colAcc0 = 0.0f, colAcc1 = 0.0f;

    for (int rowBase = rowStart; rowBase < rowEnd; rowBase += 16) {
        const int ridx = rowBase + nrel;
        const int mCl  = (ridx < rowEnd) ? ridx : (rowEnd - 1);
        // species of the row this lane loads; -1 sentinel for padded rows
        const int zq   = (ridx < rowEnd) ? zs[ridx] : -1;

        const float* arow = PS + (size_t)mCl * DIM;
        // prefetch next row-tile of A into cache (global_prefetch_b8)
        {
            int nIdx = rowBase + 16 + nrel;
            if (nIdx < rowEnd) __builtin_prefetch(PS + (size_t)nIdx * DIM, 0, 0);
        }

        // A (16x32 bf16) layout per lane: elems 0..15 <-> K = 16*hi + 0..15
        // (within each K=32 chunk, matching B).
        // Converted ONCE per wave per iteration, feeds 12 WMMAs (2 col tiles).
        v16bf ah0, al0, ah1, al1;
        load_split16(arow,      16 * hi, 16 * hi + 8, ah0, al0);    // K = 0..31
        load_split16(arow + 32, 16 * hi, 16 * hi + 8, ah1, al1);    // K = 32..63

        // G ~= Ah*Bh + Ah*Bl + Al*Bh (bf16x2 split-precision GEMM), two
        // independent accumulator chains interleaved to keep the matrix
        // pipe busy (C-chaining only; no D->A/B hazards).
        v8f acc0 = {0.f, 0.f, 0.f, 0.f, 0.f, 0.f, 0.f, 0.f};
        v8f acc1 = {0.f, 0.f, 0.f, 0.f, 0.f, 0.f, 0.f, 0.f};
        acc0 = WMMA_BF16(ah0, b0h_t0, acc0);  acc1 = WMMA_BF16(ah0, b0h_t1, acc1);
        acc0 = WMMA_BF16(ah0, b0l_t0, acc0);  acc1 = WMMA_BF16(ah0, b0l_t1, acc1);
        acc0 = WMMA_BF16(al0, b0h_t0, acc0);  acc1 = WMMA_BF16(al0, b0h_t1, acc1);
        acc0 = WMMA_BF16(ah1, b1h_t0, acc0);  acc1 = WMMA_BF16(ah1, b1h_t1, acc1);
        acc0 = WMMA_BF16(ah1, b1l_t0, acc0);  acc1 = WMMA_BF16(ah1, b1l_t1, acc1);
        acc0 = WMMA_BF16(al1, b1h_t0, acc0);  acc1 = WMMA_BF16(al1, b1h_t1, acc1);

        // C layout: VGPR r holds row (r + 8*hi) for column nrel.
        // Square, species-mask, accumulate column partials in-register.
        // z shuffles are shared across both column subtiles.
#pragma unroll
        for (int r = 0; r < 8; ++r) {
            int   zm = __shfl(zq, r + 8 * hi, 32);   // z of row rowBase + r + 8*hi
            float v0 = acc0[r];
            float v1 = acc1[r];
            colAcc0 += (zm == sc0) ? (v0 * v0) : 0.0f;
            colAcc1 += (zm == sc1) ? (v1 * v1) : 0.0f;
        }
    }

    // Combine the two lane-halves (rows 0-7 partial + rows 8-15 partial).
    colAcc0 += __shfl_xor(colAcc0, 16, 32);
    colAcc1 += __shfl_xor(colAcc1, 16, 32);

    if (lane < 16) {
        if (col0 < M) out[(size_t)b * M + col0] = colAcc0;
        if (col1 < M) out[(size_t)b * M + col1] = colAcc1;
    }
}

extern "C" void kernel_launch(void* const* d_in, const int* in_sizes, int n_in,
                              void* d_out, int out_size, void* d_ws, size_t ws_size,
                              hipStream_t stream) {
    const float* PS    = (const float*)d_in[0];
    const float* SP    = (const float*)d_in[1];
    const int*   zs    = (const int*)  d_in[2];
    const int*   batch = (const int*)  d_in[3];
    const int*   psp   = (const int*)  d_in[4];

    const int N    = in_sizes[0] / DIM;   // 200000
    const int M    = in_sizes[1] / DIM;   // 1000
    const int NMOL = out_size / M;        // 1000

    dim3 grid(NMOL, (M + COLS_PER_WG - 1) / COLS_PER_WG);
    sparse_cosine_kmat_kernel<<<grid, 128, 0, stream>>>(
        PS, SP, zs, batch, psp, (float*)d_out, N, M);
}
